// GATLayer_38405597561073
// MI455X (gfx1250) — matrix operand order repaired and checked by reading, not compiled, not run on hardware
//
#include <hip/hip_runtime.h>
#include <math.h>

#define DIM 128   // feature width D
#define KNB 30    // max neighbors K

typedef __attribute__((ext_vector_type(16))) _Float16 v16h;
typedef __attribute__((ext_vector_type(8)))  float    v8f;

__device__ __forceinline__ float lane_bcast_f(float v, int srcLane) {
  return __int_as_float(__builtin_amdgcn_readlane(__float_as_int(v), srcLane));
}

__device__ __forceinline__ float wave_max32(float v) {
  const int lane = threadIdx.x & 31;
#pragma unroll
  for (int off = 16; off > 0; off >>= 1) {
    float o = __int_as_float(
        __builtin_amdgcn_ds_bpermute((lane ^ off) << 2, __float_as_int(v)));
    v = fmaxf(v, o);
  }
  return v;
}

__device__ __forceinline__ float wave_sum32(float v) {
  const int lane = threadIdx.x & 31;
#pragma unroll
  for (int off = 16; off > 0; off >>= 1) {
    float o = __int_as_float(
        __builtin_amdgcn_ds_bpermute((lane ^ off) << 2, __float_as_int(v)));
    v += o;
  }
  return v;
}

// ---------------------------------------------------------------------------
// Stage 1: e[p] = softplus( [zf[p];zo[p]] . W )  for all P pairs.
// One wave handles 16 pairs as a WMMA tile: A = 16x256 f16 (rows = pairs),
// B = 256x16 f16 with W broadcast into every column, accumulated over 8
// chunks of K=32 with v_wmma_f32_16x16x32_f16. Every column of D equals the
// logit vector, so lanes with N==0 (lanes 0 and 16) extract 8 rows each
// using the documented C/D layout (VGPR r, lane l -> M = r + 8*(l>=16)).
// Results stored 1-based (e_ws[0] = 0 = padding slot).
// ---------------------------------------------------------------------------
__global__ void __launch_bounds__(256)
gat_logits_wmma(const float* __restrict__ zf, const float* __restrict__ zo,
                const float* __restrict__ W, float* __restrict__ e_ws,
                const int P) {
  if (blockIdx.x == 0 && threadIdx.x == 0) e_ws[0] = 0.0f;

  const int lane = threadIdx.x & 31;
  const int wid  = blockIdx.x * (blockDim.x >> 5) + (threadIdx.x >> 5);
  const int pb   = wid << 4;                 // first pair of this tile
  if (pb >= P) return;

  const int half  = lane >> 4;               // 0: lanes 0-15, 1: lanes 16-31
  const int mrow  = lane & 15;               // A-matrix row within tile
  const int prow  = min(pb + mrow, P - 1);   // clamp for ragged tail
  const size_t rb = (size_t)prow * DIM;

  v8f acc = {0.f, 0.f, 0.f, 0.f, 0.f, 0.f, 0.f, 0.f};

#pragma unroll
  for (int c = 0; c < 8; ++c) {              // K = 256 in chunks of 32
    const float* src = (c < 4) ? (zf + rb + c * 32) : (zo + rb + (c - 4) * 32);
    // 16-bit A layout, 16x32: lane<16 holds K = {0..7, 16..23}+c*32,
    // lane>=16 holds K = {8..15, 24..31}+c*32, packed low/high per VGPR.
    const float4 fa0 = *(const float4*)(src + 8 * half + 0);
    const float4 fa1 = *(const float4*)(src + 8 * half + 4);
    const float4 fb0 = *(const float4*)(src + 8 * half + 16);
    const float4 fb1 = *(const float4*)(src + 8 * half + 20);
    v16h av;
    av[0]  = (_Float16)fa0.x; av[1]  = (_Float16)fa0.y;
    av[2]  = (_Float16)fa0.z; av[3]  = (_Float16)fa0.w;
    av[4]  = (_Float16)fa1.x; av[5]  = (_Float16)fa1.y;
    av[6]  = (_Float16)fa1.z; av[7]  = (_Float16)fa1.w;
    av[8]  = (_Float16)fb0.x; av[9]  = (_Float16)fb0.y;
    av[10] = (_Float16)fb0.z; av[11] = (_Float16)fb0.w;
    av[12] = (_Float16)fb1.x; av[13] = (_Float16)fb1.y;
    av[14] = (_Float16)fb1.z; av[15] = (_Float16)fb1.w;

    // B: W broadcast across all 16 N-columns; lane<16 carries K=c*32+{0..7,16..23},
    // lane>=16 carries K=c*32+{8..15,24..31} (mirrors A's K split).
    const float* wp = W + c * 32 + 8 * half;
    const float4 w0 = *(const float4*)(wp + 0);
    const float4 w1 = *(const float4*)(wp + 4);
    const float4 w2 = *(const float4*)(wp + 16);
    const float4 w3 = *(const float4*)(wp + 20);
    v16h bv;
    bv[0]  = (_Float16)w0.x; bv[1]  = (_Float16)w0.y;
    bv[2]  = (_Float16)w0.z; bv[3]  = (_Float16)w0.w;
    bv[4]  = (_Float16)w1.x; bv[5]  = (_Float16)w1.y;
    bv[6]  = (_Float16)w1.z; bv[7]  = (_Float16)w1.w;
    bv[8]  = (_Float16)w2.x; bv[9]  = (_Float16)w2.y;
    bv[10] = (_Float16)w2.z; bv[11] = (_Float16)w2.w;
    bv[12] = (_Float16)w3.x; bv[13] = (_Float16)w3.y;
    bv[14] = (_Float16)w3.z; bv[15] = (_Float16)w3.w;

    acc = __builtin_amdgcn_wmma_f32_16x16x32_f16(
        /*neg_a=*/false, av, /*neg_b=*/false, bv,
        /*c_mod=*/(short)0, acc, /*reuse_a=*/false, /*reuse_b=*/false);
  }

  if (mrow == 0) {                            // lanes 0 and 16: column N=0
#pragma unroll
    for (int r = 0; r < 8; ++r) {
      const int p = pb + 8 * half + r;        // M = r + 8*half
      if (p < P) {
        const float x = acc[r];
        const float sp = (x > 0.0f) ? (x + log1pf(expf(-x))) : log1pf(expf(x));
        e_ws[1 + p] = sp;                     // strictly > 0 => 0 marks padding
      }
    }
  }
}

// ---------------------------------------------------------------------------
// Stage 2: one wave per node. Lane k owns neighbor k (K=30 <= 32).
// Masked softmax via ds_bpermute butterflies, then each lane accumulates 4
// of the 128 output columns over the 30 gathered 512B rows (one coalesced
// b128 load per lane per row). Neighbor rows are prefetched up front so the
// HBM gather overlaps the softmax reduction.
// ---------------------------------------------------------------------------
__global__ void __launch_bounds__(256)
gat_aggregate(const float* __restrict__ zo, const int* __restrict__ scope,
              const float* __restrict__ e_ws, float* __restrict__ out,
              const int Nn) {
  const int lane = threadIdx.x & 31;
  const int wid  = blockIdx.x * (blockDim.x >> 5) + (threadIdx.x >> 5);

  if (blockIdx.x == 0 && threadIdx.x < 32) {  // zero output row 0
    *(float4*)(out + lane * 4) = make_float4(0.f, 0.f, 0.f, 0.f);
  }
  if (wid >= Nn) return;

  int s = 0;
  if (lane < KNB) s = scope[wid * KNB + lane];   // 1-based pair index, 0 = pad

  if (s != 0) {                                  // prefetch this neighbor row
    const char* pp = (const char*)(zo + (size_t)(s - 1) * DIM);
    __builtin_prefetch(pp,       0, 3);
    __builtin_prefetch(pp + 128, 0, 3);
    __builtin_prefetch(pp + 256, 0, 3);
    __builtin_prefetch(pp + 384, 0, 3);
  }

  const float e     = (s != 0) ? e_ws[s] : 0.0f;
  const bool  valid = (e != 0.0f);
  const float logit = valid ? e : -3.402823466e+38f;   // finfo(f32).min
  const float mx    = wave_max32(logit);
  const float ex    = valid ? expf(e - mx) : 0.0f;
  float den         = wave_sum32(ex);
  den               = (den > 0.0f) ? den : 1.0f;
  const float alpha = ex / den;

  float ax = 0.f, ay = 0.f, az = 0.f, aw = 0.f;
#pragma unroll
  for (int k = 0; k < KNB; ++k) {
    const int sk = __builtin_amdgcn_readlane(s, k);  // uniform -> scalar branch
    if (sk != 0) {
      const float ak = lane_bcast_f(alpha, k);
      const float4 z = *(const float4*)(zo + (size_t)(sk - 1) * DIM + lane * 4);
      ax = fmaf(ak, z.x, ax);
      ay = fmaf(ak, z.y, ay);
      az = fmaf(ak, z.z, az);
      aw = fmaf(ak, z.w, aw);
    }
  }
  *(float4*)(out + (size_t)(wid + 1) * DIM + lane * 4) =
      make_float4(ax, ay, az, aw);
}

// ---------------------------------------------------------------------------
extern "C" void kernel_launch(void* const* d_in, const int* in_sizes, int n_in,
                              void* d_out, int out_size, void* d_ws, size_t ws_size,
                              hipStream_t stream) {
  const float* zf    = (const float*)d_in[0];   // z_feature [P,128] f32
  const float* zo    = (const float*)d_in[1];   // z_others  [P,128] f32
  const int*   scope = (const int*)  d_in[2];   // scope     [N,30]  i32
  const float* W     = (const float*)d_in[3];   // W_attn    [256,1] f32
  float*       out   = (float*)d_out;           // [N+1,128] f32
  float*       e_ws  = (float*)d_ws;            // (P+1) f32 logit table

  const int P  = in_sizes[0] / DIM;
  const int Nn = in_sizes[2] / KNB;

  const int waves1 = (P + 15) / 16;             // 16 pairs per wave
  const int grid1  = (waves1 + 7) / 8;          // 8 waves per 256-thread block
  const int grid2  = (Nn + 7) / 8;              // 1 node per wave

  gat_logits_wmma<<<grid1, 256, 0, stream>>>(zf, zo, W, e_ws, P);
  gat_aggregate <<<grid2, 256, 0, stream>>>(zo, scope, e_ws, out, Nn);
}
